// TransformerModel_29703993819860
// MI455X (gfx1250) — matrix operand order repaired and checked by reading, not compiled, not run on hardware
//
#include <hip/hip_runtime.h>
#include <hip/hip_bf16.h>
#include <math.h>

typedef __attribute__((ext_vector_type(16))) _Float16 v16h;
typedef __attribute__((ext_vector_type(8)))  _Float16 v8h;
typedef __attribute__((ext_vector_type(8)))  float    v8f;

#define S_    300
#define B_    128
#define INTOK 64
#define HID   1024
#define FIN   64
#define M_    (S_ * B_)   // 38400

enum { EP_F32 = 1, EP_F16 = 2, EP_RELU = 4 };

__device__ __forceinline__ v8f wmma16x16x32(v16h a, v16h b, v8f c) {
  // D = A(16x32 f16) * B(32x16 f16) + C(16x16 f32)
  return __builtin_amdgcn_wmma_f32_16x16x32_f16(false, a, false, b, (short)0, c,
                                                false, false);
}

// ---------------------------------------------------------------------------
// Big tiled GEMM:  C[M,N] = A[M,K] * W[N,K]^T + bias[N]   (A,W fp16; acc fp32)
// Block = 256 threads (8 wave32) arranged 4x2; block tile 256x128.
// Wave tile 64x64 = 4x4 WMMA tiles -> 16 WMMA per 16 b128 loads per k-step
// (32 flop / L1-byte).
// A-fragment per lane: row m = lane&15;
//   halves 0..7  = A[m][k0 + 16*(lane>=16) .. +7]     (VGPR0-3)
//   halves 8..15 = A[m][k0 + 8 + 16*(lane>=16) .. +7] (VGPR4-7)
// B-fragment per lane: n = lane&15, halves = W[n][k0 + 16*(lane>=16) .. +15]
// ---------------------------------------------------------------------------
template <int MODE>
__global__ __launch_bounds__(256, 1) void gemm_big_kernel(
    const _Float16* __restrict__ A, const _Float16* __restrict__ W,
    const float* __restrict__ bias, int K, int N,
    float* __restrict__ Cf, _Float16* __restrict__ Ch)
{
  const int lane = threadIdx.x & 31;
  const int wave = threadIdx.x >> 5;
  const int wm   = (wave >> 1) * 64;           // 0,64,128,192
  const int wn   = (wave & 1) * 64;            // 0,64
  const int bm   = blockIdx.x * 256;
  const int bn   = blockIdx.y * 128;
  const int lr   = lane & 15;
  const int hi   = lane >> 4;                  // 0 or 1 (half-wave)

  const _Float16* ap[4];
  const _Float16* bp[4];
#pragma unroll
  for (int t = 0; t < 4; ++t) {
    ap[t] = A + (size_t)(bm + wm + t * 16 + lr) * K + hi * 16;
    bp[t] = W + (size_t)(bn + wn + t * 16 + lr) * K + hi * 16;
  }

  v8f c[4][4];
#pragma unroll
  for (int i = 0; i < 4; ++i)
#pragma unroll
    for (int j = 0; j < 4; ++j) c[i][j] = (v8f){};

#pragma unroll 1
  for (int k0 = 0; k0 < K; k0 += 32) {
    v16h a[4], b[4];
#pragma unroll
    for (int t = 0; t < 4; ++t) {
      v8h alo = *(const v8h*)(ap[t] + k0);
      v8h ahi = *(const v8h*)(ap[t] + k0 + 8);
      a[t] = __builtin_shufflevector(alo, ahi,
                                     0,1,2,3,4,5,6,7,8,9,10,11,12,13,14,15);
      b[t] = *(const v16h*)(bp[t] + k0);
    }
#pragma unroll
    for (int i = 0; i < 4; ++i)
#pragma unroll
      for (int j = 0; j < 4; ++j)
        c[i][j] = wmma16x16x32(a[i], b[j], c[i][j]);
  }

  // Epilogue: C/D layout — VGPR r holds row (r + 8*(lane>=16)), col = lane&15
  auto store_tile = [&](const v8f& cc, int r0, int n) {
    const float bv = bias[n];
#pragma unroll
    for (int i = 0; i < 8; ++i) {
      float v = cc[i] + bv;
      if (MODE & EP_RELU) v = fmaxf(v, 0.0f);
      const size_t idx = (size_t)(r0 + i) * N + n;
      if (MODE & EP_F32) Cf[idx] = v;
      if (MODE & EP_F16) Ch[idx] = (_Float16)v;
    }
  };
#pragma unroll
  for (int i = 0; i < 4; ++i)
#pragma unroll
    for (int j = 0; j < 4; ++j)
      store_tile(c[i][j], bm + wm + i * 16 + hi * 8, bn + wn + j * 16 + lr);
}

// ---------------------------------------------------------------------------
// Small GEMM (for the final fc, N=64): block tile 128x64, wave tile 32x32.
// ---------------------------------------------------------------------------
template <int MODE>
__global__ __launch_bounds__(256) void gemm_small_kernel(
    const _Float16* __restrict__ A, const _Float16* __restrict__ W,
    const float* __restrict__ bias, int K, int N,
    float* __restrict__ Cf, _Float16* __restrict__ Ch)
{
  const int lane = threadIdx.x & 31;
  const int wave = threadIdx.x >> 5;
  const int wm   = (wave >> 1) * 32;
  const int wn   = (wave & 1) * 32;
  const int bm   = blockIdx.x * 128;
  const int bn   = blockIdx.y * 64;
  const int lr   = lane & 15;
  const int hi   = lane >> 4;

  const _Float16* a0p = A + (size_t)(bm + wm + lr) * K + hi * 16;
  const _Float16* a1p = a0p + (size_t)16 * K;
  const _Float16* b0p = W + (size_t)(bn + wn + lr) * K + hi * 16;
  const _Float16* b1p = b0p + (size_t)16 * K;

  v8f c00 = {}, c01 = {}, c10 = {}, c11 = {};

#pragma unroll 4
  for (int k0 = 0; k0 < K; k0 += 32) {
    v8h a0lo = *(const v8h*)(a0p + k0);
    v8h a0hi = *(const v8h*)(a0p + k0 + 8);
    v8h a1lo = *(const v8h*)(a1p + k0);
    v8h a1hi = *(const v8h*)(a1p + k0 + 8);
    v16h b0  = *(const v16h*)(b0p + k0);
    v16h b1  = *(const v16h*)(b1p + k0);
    v16h a0 = __builtin_shufflevector(a0lo, a0hi, 0,1,2,3,4,5,6,7,8,9,10,11,12,13,14,15);
    v16h a1 = __builtin_shufflevector(a1lo, a1hi, 0,1,2,3,4,5,6,7,8,9,10,11,12,13,14,15);
    c00 = wmma16x16x32(a0, b0, c00);
    c01 = wmma16x16x32(a0, b1, c01);
    c10 = wmma16x16x32(a1, b0, c10);
    c11 = wmma16x16x32(a1, b1, c11);
  }

  auto store_tile = [&](const v8f& c, int r0, int n) {
    const float bv = bias[n];
#pragma unroll
    for (int i = 0; i < 8; ++i) {
      float v = c[i] + bv;
      if (MODE & EP_RELU) v = fmaxf(v, 0.0f);
      const size_t idx = (size_t)(r0 + i) * N + n;
      if (MODE & EP_F32) Cf[idx] = v;
      if (MODE & EP_F16) Ch[idx] = (_Float16)v;
    }
  };
  const int rb = bm + wm + hi * 8;
  const int cb = bn + wn + lr;
  store_tile(c00, rb,      cb);
  store_tile(c01, rb,      cb + 16);
  store_tile(c10, rb + 16, cb);
  store_tile(c11, rb + 16, cb + 16);
}

// ---------------------------------------------------------------------------
// Banded attention: window j in [s-2, s]; softmax over <=3 scores.
// One wave per (s, b, head); HD=256 -> 8 elems/lane; shuffle-reduce dots.
// ---------------------------------------------------------------------------
__global__ __launch_bounds__(256) void attn_kernel(const _Float16* __restrict__ qkv,
                                                   _Float16* __restrict__ o)
{
  const int lane = threadIdx.x & 31;
  const int wid  = blockIdx.x * 8 + (threadIdx.x >> 5);
  const int h = wid & 3;
  const int b = (wid >> 2) & (B_ - 1);
  const int s = wid >> 9;            // / (4*128)
  if (s >= S_) return;
  const int d0 = lane * 8;

  const size_t qoff = (size_t)(s * B_ + b) * (3 * HID) + h * 256 + d0;
  float qv[8];
#pragma unroll
  for (int i = 0; i < 8; ++i) qv[i] = (float)qkv[qoff + i];

  const int jmin = (s >= 2) ? (s - 2) : 0;
  const int cnt  = s - jmin + 1;
  float sc[3];
  float mx = -1e30f;
  for (int t = 0; t < cnt; ++t) {
    const int j = jmin + t;
    const size_t koff = (size_t)(j * B_ + b) * (3 * HID) + HID + h * 256 + d0;
    float dot = 0.0f;
#pragma unroll
    for (int i = 0; i < 8; ++i) dot += qv[i] * (float)qkv[koff + i];
#pragma unroll
    for (int off = 16; off >= 1; off >>= 1) dot += __shfl_xor(dot, off, 32);
    sc[t] = dot * 0.0625f;           // 1/sqrt(256)
    mx = fmaxf(mx, sc[t]);
  }
  float wexp[3], sum = 0.0f;
  for (int t = 0; t < cnt; ++t) { wexp[t] = expf(sc[t] - mx); sum += wexp[t]; }
  const float inv = 1.0f / sum;

  float acc[8] = {0, 0, 0, 0, 0, 0, 0, 0};
  for (int t = 0; t < cnt; ++t) {
    const int j = jmin + t;
    const size_t voff = (size_t)(j * B_ + b) * (3 * HID) + 2 * HID + h * 256 + d0;
    const float wt = wexp[t] * inv;
#pragma unroll
    for (int i = 0; i < 8; ++i) acc[i] += wt * (float)qkv[voff + i];
  }
  _Float16* op = o + (size_t)(s * B_ + b) * HID + h * 256 + d0;
#pragma unroll
  for (int i = 0; i < 8; ++i) op[i] = (_Float16)acc[i];
}

// ---------------------------------------------------------------------------
// Fused residual + LayerNorm over rows of 1024. One 256-thread block per row.
// ---------------------------------------------------------------------------
__global__ __launch_bounds__(256) void ln_kernel(const float* __restrict__ xin,
                                                 const float* __restrict__ y,
                                                 const float* __restrict__ g,
                                                 const float* __restrict__ beta,
                                                 float* __restrict__ xout,
                                                 _Float16* __restrict__ xh)
{
  const int row = blockIdx.x;
  const int tid = threadIdx.x;
  const size_t base = (size_t)row * HID + tid * 4;
  float v[4];
  float s = 0.0f, sq = 0.0f;
#pragma unroll
  for (int i = 0; i < 4; ++i) {
    v[i] = xin[base + i] + y[base + i];
    s += v[i]; sq += v[i] * v[i];
  }
#pragma unroll
  for (int off = 16; off >= 1; off >>= 1) {
    s  += __shfl_xor(s,  off, 32);
    sq += __shfl_xor(sq, off, 32);
  }
  __shared__ float shs[8], shq[8];
  if ((tid & 31) == 0) { shs[tid >> 5] = s; shq[tid >> 5] = sq; }
  __syncthreads();
  float ts = 0.0f, tq = 0.0f;
#pragma unroll
  for (int w = 0; w < 8; ++w) { ts += shs[w]; tq += shq[w]; }
  const float mu  = ts * (1.0f / HID);
  const float var = tq * (1.0f / HID) - mu * mu;
  const float inv = rsqrtf(var + 1e-5f);
#pragma unroll
  for (int i = 0; i < 4; ++i) {
    const int c = tid * 4 + i;
    const float o = (v[i] - mu) * inv * g[c] + beta[c];
    xout[base + i] = o;
    xh[base + i]   = (_Float16)o;
  }
}

__global__ void f32_to_f16_kernel(const float* __restrict__ in,
                                  _Float16* __restrict__ out, size_t n)
{
  const size_t i = ((size_t)blockIdx.x * blockDim.x + threadIdx.x) * 4;
  if (i < n) {
#pragma unroll
    for (int k = 0; k < 4; ++k) out[i + k] = (_Float16)in[i + k];
  }
}

// ---------------------------------------------------------------------------
extern "C" void kernel_launch(void* const* d_in, const int* in_sizes, int n_in,
                              void* d_out, int out_size, void* d_ws, size_t ws_size,
                              hipStream_t stream) {
  (void)in_sizes; (void)n_in; (void)out_size; (void)ws_size;
  const float* src        = (const float*)d_in[0];
  const float* enc_w      = (const float*)d_in[1];
  const float* enc_b      = (const float*)d_in[2];
  const float* in_proj_w  = (const float*)d_in[3];
  const float* in_proj_b  = (const float*)d_in[4];
  const float* out_proj_w = (const float*)d_in[5];
  const float* out_proj_b = (const float*)d_in[6];
  const float* ln1_g      = (const float*)d_in[7];
  const float* ln1_b      = (const float*)d_in[8];
  const float* lin1_w     = (const float*)d_in[9];
  const float* lin1_b     = (const float*)d_in[10];
  const float* lin2_w     = (const float*)d_in[11];
  const float* lin2_b     = (const float*)d_in[12];
  const float* ln2_g      = (const float*)d_in[13];
  const float* ln2_b      = (const float*)d_in[14];
  const float* fc_w       = (const float*)d_in[15];
  const float* fc_b       = (const float*)d_in[16];

  // Workspace carve-out
  char* p = (char*)d_ws;
  auto carve = [&](size_t bytes) -> void* {
    void* r = (void*)p;
    p += (bytes + 255) & ~(size_t)255;
    return r;
  };
  _Float16* xh     = (_Float16*)carve((size_t)M_ * HID * 2);
  float*    x      = (float*)   carve((size_t)M_ * HID * 4);
  void*     shrd   =            carve((size_t)M_ * 3 * HID * 2); // qkvh aliases y
  _Float16* qkvh   = (_Float16*)shrd;
  float*    y      = (float*)shrd;
  _Float16* oh     = (_Float16*)carve((size_t)M_ * HID * 2);     // also FF hidden
  _Float16* srch   = (_Float16*)carve((size_t)M_ * INTOK * 2);
  _Float16* enc_wh = (_Float16*)carve((size_t)HID * INTOK * 2);
  _Float16* ipw_h  = (_Float16*)carve((size_t)3 * 3 * HID * HID * 2);
  _Float16* opw_h  = (_Float16*)carve((size_t)3 * HID * HID * 2);
  _Float16* l1w_h  = (_Float16*)carve((size_t)3 * HID * HID * 2);
  _Float16* l2w_h  = (_Float16*)carve((size_t)3 * HID * HID * 2);
  _Float16* fcw_h  = (_Float16*)carve((size_t)FIN * HID * 2);

  auto conv = [&](const float* a, _Float16* b, size_t n) {
    f32_to_f16_kernel<<<(unsigned)((n + 1023) / 1024), 256, 0, stream>>>(a, b, n);
  };
  conv(src,        srch,   (size_t)M_ * INTOK);
  conv(enc_w,      enc_wh, (size_t)HID * INTOK);
  conv(in_proj_w,  ipw_h,  (size_t)3 * 3 * HID * HID);
  conv(out_proj_w, opw_h,  (size_t)3 * HID * HID);
  conv(lin1_w,     l1w_h,  (size_t)3 * HID * HID);
  conv(lin2_w,     l2w_h,  (size_t)3 * HID * HID);
  conv(fc_w,       fcw_h,  (size_t)FIN * HID);

  const dim3 blk(256);
  const dim3 gN1024(M_ / 256, HID / 128);       // 150 x 8
  const dim3 gN3072(M_ / 256, (3 * HID) / 128); // 150 x 24

  // x = src @ enc_w^T + enc_b   (writes fp32 x and fp16 xh)
  gemm_big_kernel<EP_F32 | EP_F16>
      <<<gN1024, blk, 0, stream>>>(srch, enc_wh, enc_b, INTOK, HID, x, xh);

  for (int l = 0; l < 3; ++l) {
    // qkv = xh @ in_proj_w[l]^T + b  -> fp16
    gemm_big_kernel<EP_F16>
        <<<gN3072, blk, 0, stream>>>(
            xh, ipw_h + (size_t)l * 3 * HID * HID, in_proj_b + l * 3 * HID,
            HID, 3 * HID, (float*)nullptr, qkvh);

    // banded self-attention (window of 3)
    attn_kernel<<<(S_ * B_ * 4) / 8, blk, 0, stream>>>(qkvh, oh);

    // y = o @ out_proj_w[l]^T + b  -> fp32 (overwrites qkv region)
    gemm_big_kernel<EP_F32>
        <<<gN1024, blk, 0, stream>>>(
            oh, opw_h + (size_t)l * HID * HID, out_proj_b + l * HID,
            HID, HID, y, (_Float16*)nullptr);

    // x = LN(x + y)
    ln_kernel<<<M_, blk, 0, stream>>>(x, y, ln1_g + l * HID, ln1_b + l * HID, x, xh);

    // ff = relu(xh @ lin1_w^T + b) -> fp16 (reuse oh)
    gemm_big_kernel<EP_F16 | EP_RELU>
        <<<gN1024, blk, 0, stream>>>(
            xh, l1w_h + (size_t)l * HID * HID, lin1_b + l * HID,
            HID, HID, (float*)nullptr, oh);

    // y = ff @ lin2_w^T + b -> fp32
    gemm_big_kernel<EP_F32>
        <<<gN1024, blk, 0, stream>>>(
            oh, l2w_h + (size_t)l * HID * HID, lin2_b + l * HID,
            HID, HID, y, (_Float16*)nullptr);

    // x = LN(x + y)
    ln_kernel<<<M_, blk, 0, stream>>>(x, y, ln2_g + l * HID, ln2_b + l * HID, x, xh);
  }

  // out = xh @ fc_w^T + fc_b  -> fp32 [38400, 64]
  gemm_small_kernel<EP_F32>
      <<<dim3(M_ / 128, FIN / 64), blk, 0, stream>>>(xh, fcw_h, fc_b,
                                                     HID, FIN, (float*)d_out,
                                                     (_Float16*)nullptr);
}
